// PhiFFN_56650618634409
// MI455X (gfx1250) — compile-verified, matrix operand order checked by reading, not run on hardware
//
#include <hip/hip_runtime.h>

// ---------------------------------------------------------------------------
// out = ALPHA*x + BETA*(x @ W^T),  W densified from COO (duplicates sum).
// Pipeline:
//   1) zero dense W (fp32) in workspace
//   2) scatter-add COO entries (global_atomic_add_f32)
//   3) pack W^T into split-fp16 (hi/lo) WMMA B-fragment layout in workspace
//   4) GEMM: A tile staged via LDS (fp32 -> f16 hi/lo split ONCE per block,
//      fragment order, double-buffered; guard-free fast path for full tiles),
//      B fragments direct from L2-resident packed buffers, 3-product
//      split-f16 WMMA accumulation, fused residual epilogue.
// ---------------------------------------------------------------------------

#define DIM       1910
#define KT_COUNT  60          // ceil(1910/32) -> K padded to 1920
#define NT_COUNT  120         // ceil(1910/16) -> N padded to 1920
#define NBLOCKS   15          // ceil(1910/128)

static constexpr float ALPHA = 0.6180339887498949f;   // 1/phi
static constexpr float BETA  = 0.3819660112501051f;   // 1/phi^2

typedef __attribute__((ext_vector_type(16))) _Float16 v16h;
typedef __attribute__((ext_vector_type(2)))  _Float16 h2;
typedef __attribute__((ext_vector_type(8)))  float    v8f;

// ---------------------------------------------------------------- zero dense W
__global__ void PhiFFN_zero_kernel(float* __restrict__ W, int n) {
    int i = blockIdx.x * blockDim.x + threadIdx.x;
    if (i < n) W[i] = 0.0f;
}

// ------------------------------------------------------------- COO scatter-add
__global__ void PhiFFN_scatter_kernel(const int* __restrict__ rows,
                                      const int* __restrict__ cols,
                                      const float* __restrict__ vals,
                                      float* __restrict__ W, int nz) {
    int i = blockIdx.x * blockDim.x + threadIdx.x;
    if (i < nz) {
        atomicAdd(&W[(long)rows[i] * DIM + cols[i]], vals[i]);
    }
}

// ------------------------------------------------- pack B = W^T into fragments
// B fragment layout (16-bit 32x16 KxN tile, wave32), mirrored from the ISA
// A-matrix 16x32 table: lane = n_local + 16*s; half h holds
// k_local = (h&7) + 8*s + 16*(h>>3).  One thread packs one (tile, lane) slot:
// a contiguous 32-byte chunk, so the GEMM loads it with plain b128 loads.
__global__ void PhiFFN_packb_kernel(const float* __restrict__ W,
                                    _Float16* __restrict__ Bhi,
                                    _Float16* __restrict__ Blo) {
    int tid = blockIdx.x * blockDim.x + threadIdx.x;
    const int total = KT_COUNT * NT_COUNT * 32;
    if (tid >= total) return;
    int lane = tid & 31;
    int tile = tid >> 5;
    int nt   = tile % NT_COUNT;
    int kt   = tile / NT_COUNT;
    int s    = lane >> 4;
    int n    = nt * 16 + (lane & 15);

    v16h hi = {};
    v16h lo = {};
#pragma unroll
    for (int h = 0; h < 16; ++h) {
        int k = kt * 32 + (h & 7) + 8 * s + 16 * (h >> 3);
        float w = 0.0f;
        if (n < DIM && k < DIM) w = W[(long)n * DIM + k];   // B[k][n] = W[n][k]
        _Float16 whi = (_Float16)w;
        hi[h] = whi;
        lo[h] = (_Float16)(w - (float)whi);
    }
    long off = (long)tile * 512 + (long)lane * 16;          // halves
    *reinterpret_cast<v16h*>(Bhi + off) = hi;
    *reinterpret_cast<v16h*>(Blo + off) = lo;
}

// -------------------------------------------------- cooperative A-tile staging
// 128 rows x 32 k of fp32 x -> split f16 hi/lo, written to LDS in WMMA
// A-fragment order.  q enumerates k-pairs row-major so global float2 reads
// coalesce; each pair lands as one dword (ds_store_b32) of a fragment:
//   sm = m_local>>4, lane = (m_local&15) | (((k&8)!=0)<<4),
//   dword j = ((k&7)>>1) | ((k>>4)<<2)     (k even).
// FULL=true: tile entirely in range -> no per-element guards (the common case:
// 59 of 60 K-tiles, all M-bands since tokens % 128 == 0).
template <bool FULL>
__device__ __forceinline__
void PhiFFN_stage_a(const float* __restrict__ x, int tokens, int mrow0, int k0,
                    int tid, _Float16* __restrict__ hi, _Float16* __restrict__ lo) {
#pragma unroll
    for (int r = 0; r < 8; ++r) {
        int q       = r * 256 + tid;        // 2048 k-pairs per tile
        int m_local = q >> 4;
        int k_local = (q & 15) << 1;
        int m  = mrow0 + m_local;
        int kg = k0 + k_local;
        float w0, w1;
        if (FULL) {
            float2 t = *reinterpret_cast<const float2*>(x + (long)m * DIM + kg);
            w0 = t.x; w1 = t.y;
        } else {
            w0 = 0.0f; w1 = 0.0f;
            if (m < tokens) {
                const float* xrow = x + (long)m * DIM;
                if (kg + 1 < DIM) {
                    float2 t = *reinterpret_cast<const float2*>(xrow + kg);
                    w0 = t.x; w1 = t.y;
                } else if (kg < DIM) {
                    w0 = xrow[kg];
                }
            }
        }
        _Float16 h0 = (_Float16)w0, h1 = (_Float16)w1;
        _Float16 l0 = (_Float16)(w0 - (float)h0);
        _Float16 l1 = (_Float16)(w1 - (float)h1);
        int sm   = m_local >> 4;
        int lane = (m_local & 15) | (((k_local >> 3) & 1) << 4);
        int j    = ((k_local & 7) >> 1) | ((k_local >> 4) << 2);
        int d    = (sm * 32 + lane) * 8 + j;
        *reinterpret_cast<h2*>(hi + d * 2) = (h2){h0, h1};
        *reinterpret_cast<h2*>(lo + d * 2) = (h2){l0, l1};
    }
}

// ----------------------------------------------------------------------- GEMM
// Block: 256 threads = 8 waves, tile 128(M) x 128(N).
// Wave grid 2(M) x 4(N): each wave owns 64x32 = 4x2 accumulators of 16x16.
// grid.x = N-blocks (15), grid.y = M-blocks -> N varies fastest so the current
// 128-row band of x stays hot in L2 across its 15 N-blocks.
__global__ __launch_bounds__(256, 1)
void PhiFFN_gemm_kernel(const float* __restrict__ x,
                        const _Float16* __restrict__ Bhi,
                        const _Float16* __restrict__ Blo,
                        float* __restrict__ out, int tokens) {
    __shared__ __align__(32) _Float16 sAhi[2][8 * 32 * 16];   // 2 x 8 KB
    __shared__ __align__(32) _Float16 sAlo[2][8 * 32 * 16];   // 2 x 8 KB

    const int tid    = threadIdx.x;
    const int lane   = tid & 31;
    const int wid    = tid >> 5;
    const int waveM  = wid & 1;          // 0..1  (64 rows each)
    const int waveN  = wid >> 1;         // 0..3  (32 cols each)
    const int s      = lane >> 4;
    const int lane_m = lane & 15;

    const int mrow0     = blockIdx.y * 128;
    const int mrow_base = mrow0 + waveM * 64;
    const int nt_base   = blockIdx.x * 8 + waveN * 2;   // 16-wide N subtiles
    const bool mfull    = (mrow0 + 128 <= tokens);      // block-uniform

    v8f acc[4][2];
#pragma unroll
    for (int mi = 0; mi < 4; ++mi)
#pragma unroll
        for (int ni = 0; ni < 2; ++ni)
            acc[mi][ni] = (v8f){};

    // prologue: stage first K-tile (k0=0, always K-full)
    if (mfull) PhiFFN_stage_a<true >(x, tokens, mrow0, 0, tid, sAhi[0], sAlo[0]);
    else       PhiFFN_stage_a<false>(x, tokens, mrow0, 0, tid, sAhi[0], sAlo[0]);
    __syncthreads();

    for (int kt = 0; kt < KT_COUNT; ++kt) {
        const int cur = kt & 1;

        // stage next K-tile into the other buffer (overlaps with WMMAs below);
        // block-uniform scalar branch picks the guard-free path.
        if (kt + 1 < KT_COUNT) {
            const int k0n = (kt + 1) * 32;
            if (mfull && (k0n + 32 <= DIM))
                PhiFFN_stage_a<true >(x, tokens, mrow0, k0n, tid,
                                      sAhi[cur ^ 1], sAlo[cur ^ 1]);
            else
                PhiFFN_stage_a<false>(x, tokens, mrow0, k0n, tid,
                                      sAhi[cur ^ 1], sAlo[cur ^ 1]);
        }

        // ---- B fragments: contiguous 32B per lane, straight from global (L2)
        v16h bhi[2], blo[2];
#pragma unroll
        for (int ni = 0; ni < 2; ++ni) {
            long boff = ((long)(kt * NT_COUNT + nt_base + ni) * 32 + lane) * 16;
            bhi[ni] = *reinterpret_cast<const v16h*>(Bhi + boff);
            blo[ni] = *reinterpret_cast<const v16h*>(Blo + boff);
        }
        if (kt + 1 < KT_COUNT) {   // hint next B tile toward L2/L0
            long pboff = ((long)((kt + 1) * NT_COUNT + nt_base) * 32 + lane) * 16;
            __builtin_prefetch(Bhi + pboff, 0, 1);
            __builtin_prefetch(Blo + pboff, 0, 1);
        }

        // ---- A fragments: contiguous 32B per lane from LDS (ds_load_b128 x2)
        v16h ahi[4], alo[4];
#pragma unroll
        for (int mi = 0; mi < 4; ++mi) {
            int base = ((waveM * 4 + mi) * 32 + lane) * 16;
            ahi[mi] = *reinterpret_cast<const v16h*>(&sAhi[cur][base]);
            alo[mi] = *reinterpret_cast<const v16h*>(&sAlo[cur][base]);
        }

        // ---- 3-product split-fp16 accumulation (hi*hi + hi*lo + lo*hi)
#pragma unroll
        for (int mi = 0; mi < 4; ++mi) {
#pragma unroll
            for (int ni = 0; ni < 2; ++ni) {
                acc[mi][ni] = __builtin_amdgcn_wmma_f32_16x16x32_f16(
                    false, ahi[mi], false, bhi[ni], (short)0, acc[mi][ni], false, false);
                acc[mi][ni] = __builtin_amdgcn_wmma_f32_16x16x32_f16(
                    false, ahi[mi], false, blo[ni], (short)0, acc[mi][ni], false, false);
                acc[mi][ni] = __builtin_amdgcn_wmma_f32_16x16x32_f16(
                    false, alo[mi], false, bhi[ni], (short)0, acc[mi][ni], false, false);
            }
        }

        __syncthreads();   // one barrier per iter (double-buffered staging)
    }

    // ---- epilogue: out = ALPHA*x + BETA*acc
    // C/D layout: VGPR v -> M = v + 8*s, N = lane&15.
    // Uniform dispatch removes the per-row m-guard in the common case.
    auto epilogue = [&](auto mguard) {
#pragma unroll
        for (int mi = 0; mi < 4; ++mi) {
#pragma unroll
            for (int ni = 0; ni < 2; ++ni) {
                int n = blockIdx.x * 128 + waveN * 32 + ni * 16 + lane_m;
                if (n >= DIM) continue;
#pragma unroll
                for (int v = 0; v < 8; ++v) {
                    int m = mrow_base + mi * 16 + v + 8 * s;
                    if (!mguard.value || m < tokens) {
                        long idx = (long)m * DIM + n;
                        out[idx] = ALPHA * x[idx] + BETA * acc[mi][ni][v];
                    }
                }
            }
        }
    };
    if (mfull) epilogue(std::integral_constant<bool, false>{});   // no m-guard
    else       epilogue(std::integral_constant<bool, true >{});   // guarded

}

// ---------------------------------------------------------------------- launch
extern "C" void kernel_launch(void* const* d_in, const int* in_sizes, int n_in,
                              void* d_out, int out_size, void* d_ws, size_t ws_size,
                              hipStream_t stream) {
    const float* x    = (const float*)d_in[0];
    const int*   Widx = (const int*)d_in[1];     // (2, NZ): rows then cols
    const float* Wval = (const float*)d_in[2];
    float* out = (float*)d_out;

    const int  nz     = in_sizes[2];
    const long tokens = (long)in_sizes[0] / DIM;

    // Workspace layout: [dense W fp32 | Bhi f16 | Blo f16]  (~29.3 MB)
    char* ws = (char*)d_ws;
    size_t wBytes = ((size_t)DIM * DIM * sizeof(float) + 511) & ~(size_t)511;
    float*    Wd  = (float*)ws;
    size_t    bEl = (size_t)KT_COUNT * NT_COUNT * 512;   // halves per buffer
    _Float16* Bhi = (_Float16*)(ws + wBytes);
    _Float16* Blo = Bhi + bEl;

    const int wn = DIM * DIM;
    PhiFFN_zero_kernel<<<(wn + 255) / 256, 256, 0, stream>>>(Wd, wn);
    PhiFFN_scatter_kernel<<<(nz + 255) / 256, 256, 0, stream>>>(
        Widx, Widx + nz, Wval, Wd, nz);
    const int pt = KT_COUNT * NT_COUNT * 32;
    PhiFFN_packb_kernel<<<(pt + 255) / 256, 256, 0, stream>>>(Wd, Bhi, Blo);

    dim3 grid(NBLOCKS, (unsigned)((tokens + 127) / 128));
    PhiFFN_gemm_kernel<<<grid, 256, 0, stream>>>(x, Bhi, Blo, out, (int)tokens);
}
